// Seq2SeqLSTM_Attn_31404800868670
// MI455X (gfx1250) — matrix operand order repaired and checked by reading, not compiled, not run on hardware
//
#include <hip/hip_runtime.h>
#include <hip/hip_bf16.h>
#include <math.h>

// Problem constants (match reference)
#define VOCAB  32000
#define EDIM   512
#define HDIM   1024
#define GDIM   4096          // 4*H
#define BATCH  32
#define SLEN   64
#define DSTEPS 63            // T-1
#define DECIN  (EDIM + HDIM) // 1536
#define KSLICES 2            // K-split for the gates GEMM

typedef __bf16 bf16_t;
typedef __attribute__((ext_vector_type(16))) __bf16 v16bf;
typedef __attribute__((ext_vector_type(8)))  __bf16 v8bf;
typedef __attribute__((ext_vector_type(8)))  float  v8f;

// ---------------------------------------------------------------------------
// WMMA fragment loaders (v_wmma_f32_16x16x32_bf16 layouts, wave32)
// A 16x32 bf16 (row-major, row stride in elements):
//   lane = half*16 + m ; element i -> K = (i/8)*16 + half*8 + (i%8)
// B 32x16 bf16 where B[k][n] = W[n][k], W row-major (N x K):
//   lane = half*16 + n ; element i -> K = (i/8)*16 + half*8 + (i%8)
// C/D 16x16 f32: lane = half*16 + n ; acc[r] -> M = half*8 + r
// ---------------------------------------------------------------------------
__device__ __forceinline__ v16bf load_a_frag(const bf16_t* A, size_t row_stride,
                                             int m0, int k0, int lane) {
  const int half = (lane >> 4) & 1, ml = lane & 15;
  const bf16_t* p = A + (size_t)(m0 + ml) * row_stride + k0 + half * 8;
  v8bf lo = *(const v8bf*)p;          // K = half*8 .. half*8+7
  v8bf hi = *(const v8bf*)(p + 16);   // K = 16+half*8 .. 16+half*8+7
  v16bf r;
#pragma unroll
  for (int i = 0; i < 8; ++i) { r[i] = lo[i]; r[i + 8] = hi[i]; }
  return r;
}

__device__ __forceinline__ v16bf load_b_frag(const bf16_t* W, size_t row_stride,
                                             int n0, int k0, int lane) {
  const int half = (lane >> 4) & 1, nl = lane & 15;
  const bf16_t* p = W + (size_t)(n0 + nl) * row_stride + k0 + half * 8;
  v8bf lo = *(const v8bf*)p;
  v8bf hi = *(const v8bf*)(p + 16);
  v16bf r;
#pragma unroll
  for (int i = 0; i < 8; ++i) { r[i] = lo[i]; r[i + 8] = hi[i]; }
  return r;
}

#define WMMA_BF16(a, b, c) \
  __builtin_amdgcn_wmma_f32_16x16x32_bf16(false, (a), false, (b), (short)0, (c), false, false)

// ---------------------------------------------------------------------------
// Gates GEMM, K-sliced: partial[ks][B x 4H] = X[:, ksliceKx] @ Wih^T slice
//                                           + Hprev[:, ksliceH] @ Whh^T slice
// Each wave owns one 16-col N-tile and BOTH 16-row M-tiles: B fragment is
// loaded once per K-chunk and feeds two independent WMMA accumulator chains.
// 256 N-tiles x KSLICES = 512 waves.
// ---------------------------------------------------------------------------
__global__ void lstm_gates_kernel(const bf16_t* __restrict__ X, size_t x_row_stride, int Kx,
                                  const bf16_t* __restrict__ Wih,
                                  const bf16_t* __restrict__ Hprev,
                                  const bf16_t* __restrict__ Whh,
                                  float* __restrict__ partial) {
  const int lane  = threadIdx.x & 31;
  const int wave  = blockIdx.x * (blockDim.x >> 5) + (threadIdx.x >> 5);
  const int tileN = wave % (GDIM / 16);
  const int ks    = wave / (GDIM / 16);     // k-slice index
  const int n0 = tileN * 16;
  v8f acc0 = {}, acc1 = {};                 // m-tile 0 (rows 0-15), 1 (rows 16-31)

  const int kxw = Kx / KSLICES;             // multiple of 32 for Kx in {512,1024,1536}
  const int kxb = ks * kxw, kxe = kxb + kxw;
  for (int k0 = kxb; k0 < kxe; k0 += 32) {
    v16bf b  = load_b_frag(Wih, (size_t)Kx, n0, k0, lane);
    v16bf a0 = load_a_frag(X, x_row_stride, 0,  k0, lane);
    v16bf a1 = load_a_frag(X, x_row_stride, 16, k0, lane);
    acc0 = WMMA_BF16(a0, b, acc0);
    acc1 = WMMA_BF16(a1, b, acc1);
  }
  const int khw = HDIM / KSLICES;
  const int khb = ks * khw, khe = khb + khw;
  for (int k0 = khb; k0 < khe; k0 += 32) {
    v16bf b  = load_b_frag(Whh, HDIM, n0, k0, lane);
    v16bf a0 = load_a_frag(Hprev, HDIM, 0,  k0, lane);
    v16bf a1 = load_a_frag(Hprev, HDIM, 16, k0, lane);
    acc0 = WMMA_BF16(a0, b, acc0);
    acc1 = WMMA_BF16(a1, b, acc1);
  }
  const int nl = lane & 15, half = lane >> 4;
  float* dst = partial + (size_t)ks * BATCH * GDIM;
#pragma unroll
  for (int r = 0; r < 8; ++r) {
    int m = half * 8 + r;
    dst[(size_t)m * GDIM + n0 + nl]        = acc0[r];
    dst[(size_t)(m + 16) * GDIM + n0 + nl] = acc1[r];
  }
}

// ---------------------------------------------------------------------------
// out[b, t, :] = h[b,:] @ fcW^T + fc_b     (N = 32000, K = 1024)
// One N-tile per wave, both M-tiles per wave (B loaded once, 2 acc chains).
// 2000 waves; fcW (64MB bf16) stays L2-resident across all 63 steps.
// ---------------------------------------------------------------------------
__global__ void fc_kernel(const bf16_t* __restrict__ Hd,
                          const bf16_t* __restrict__ W,
                          const float* __restrict__ bias,
                          float* __restrict__ out, int t) {
  const int lane = threadIdx.x & 31;
  const int wave = blockIdx.x * (blockDim.x >> 5) + (threadIdx.x >> 5);
  const int n0 = wave * 16;                 // 2000 N-tiles
  v8f acc0 = {}, acc1 = {};
  for (int k0 = 0; k0 < HDIM; k0 += 32) {
    v16bf b  = load_b_frag(W, HDIM, n0, k0, lane);
    v16bf a0 = load_a_frag(Hd, HDIM, 0,  k0, lane);
    v16bf a1 = load_a_frag(Hd, HDIM, 16, k0, lane);
    acc0 = WMMA_BF16(a0, b, acc0);
    acc1 = WMMA_BF16(a1, b, acc1);
  }
  const int nl = lane & 15, half = lane >> 4;
  const float bv = bias[n0 + nl];
#pragma unroll
  for (int r = 0; r < 8; ++r) {
    int m = half * 8 + r;                   // batch row in tile 0
    out[((size_t)m * DSTEPS + t) * VOCAB + n0 + nl]        = acc0[r] + bv;
    out[((size_t)(m + 16) * DSTEPS + t) * VOCAB + n0 + nl] = acc1[r] + bv;
  }
}

// ---------------------------------------------------------------------------
// Elementwise LSTM cell. Sums KSLICES partial gate buffers, adds biases,
// applies sigmoid/tanh; c kept f32, h emitted bf16 (+ optional copies).
// ---------------------------------------------------------------------------
__global__ void lstm_pointwise_kernel(const float* __restrict__ partial,
                                      const float* __restrict__ bih,
                                      const float* __restrict__ bhh,
                                      float* __restrict__ c,
                                      float* __restrict__ h_f32,   // nullable
                                      bf16_t* __restrict__ h_bf,
                                      bf16_t* __restrict__ seq_bf, // nullable, pre-offset by t*H
                                      float* __restrict__ seq_f32) // nullable, pre-offset by t*H
{
  int idx = blockIdx.x * blockDim.x + threadIdx.x;
  if (idx >= BATCH * HDIM) return;
  int b = idx >> 10, j = idx & (HDIM - 1);
  const float* g0 = partial + (size_t)b * GDIM;
  const float* g1 = g0 + (size_t)BATCH * GDIM;
  float gi = g0[j]            + g1[j]            + bih[j]            + bhh[j];
  float gf = g0[HDIM + j]     + g1[HDIM + j]     + bih[HDIM + j]     + bhh[HDIM + j];
  float gg = g0[2 * HDIM + j] + g1[2 * HDIM + j] + bih[2 * HDIM + j] + bhh[2 * HDIM + j];
  float go = g0[3 * HDIM + j] + g1[3 * HDIM + j] + bih[3 * HDIM + j] + bhh[3 * HDIM + j];
  float si = 1.0f / (1.0f + expf(-gi));
  float sf = 1.0f / (1.0f + expf(-gf));
  float so = 1.0f / (1.0f + expf(-go));
  float cn = sf * c[idx] + si * tanhf(gg);
  c[idx] = cn;
  float hv = so * tanhf(cn);
  h_bf[idx] = (bf16_t)hv;
  if (h_f32)  h_f32[idx] = hv;
  if (seq_bf)  seq_bf[(size_t)b * (SLEN * HDIM) + j]  = (bf16_t)hv;
  if (seq_f32) seq_f32[(size_t)b * (SLEN * HDIM) + j] = hv;
}

// ---------------------------------------------------------------------------
// Attention + decoder-input assembly (tiny: ~8 MFLOP/step, f32 VALU).
// One block per batch row; 4 threads cooperate per score.
// Writes dec_in = [emb_t | context] as bf16.
// ---------------------------------------------------------------------------
__global__ void attn_kernel(const float* __restrict__ enc_out,  // (B,S,H) f32
                            const float* __restrict__ hq,       // (B,H)  f32
                            const bf16_t* __restrict__ dec_emb, // (B,63,E) bf16
                            int t,
                            bf16_t* __restrict__ dec_in) {      // (B, E+H) bf16
  const int b = blockIdx.x;
  const int tid = threadIdx.x;
  __shared__ float part[SLEN][4];
  __shared__ float sc[SLEN];
  const float* eb = enc_out + (size_t)b * SLEN * HDIM;
  const float* hb = hq + (size_t)b * HDIM;
  {
    const int s = tid >> 2, q = tid & 3;    // 64 scores x 4 threads
    float acc = 0.0f;
    for (int k = q; k < HDIM; k += 4) acc += eb[(size_t)s * HDIM + k] * hb[k];
    part[s][q] = acc;
  }
  __syncthreads();
  if (tid < SLEN)
    sc[tid] = part[tid][0] + part[tid][1] + part[tid][2] + part[tid][3];
  __syncthreads();
  if (tid == 0) {
    float m = sc[0];
    for (int s = 1; s < SLEN; ++s) m = fmaxf(m, sc[s]);
    float sum = 0.0f;
    for (int s = 0; s < SLEN; ++s) { sc[s] = expf(sc[s] - m); sum += sc[s]; }
    float inv = 1.0f / sum;
    for (int s = 0; s < SLEN; ++s) sc[s] *= inv;
  }
  __syncthreads();
  bf16_t* din = dec_in + (size_t)b * DECIN;
  for (int h = tid; h < HDIM; h += blockDim.x) {
    float ctx = 0.0f;
    for (int s = 0; s < SLEN; ++s) ctx += sc[s] * eb[(size_t)s * HDIM + h];
    din[EDIM + h] = (bf16_t)ctx;
  }
  for (int j = tid; j < EDIM; j += blockDim.x)
    din[j] = dec_emb[((size_t)b * DSTEPS + t) * EDIM + j];
}

// ---------------------------------------------------------------------------
// Embedding lookup -> bf16, with PAD_ID(=0) row zeroed. Token row stride = 64
// for both src (all 64 cols) and tgt (first 63 cols).
// ---------------------------------------------------------------------------
__global__ void embed_lookup_kernel(const int* __restrict__ tokens, int tok_cols,
                                    const float* __restrict__ embed,
                                    bf16_t* __restrict__ out, size_t total) {
  for (size_t idx = blockIdx.x * (size_t)blockDim.x + threadIdx.x; idx < total;
       idx += (size_t)gridDim.x * blockDim.x) {
    size_t row = idx / EDIM;
    int e = (int)(idx % EDIM);
    int b = (int)(row / tok_cols), p = (int)(row % tok_cols);
    int tok = tokens[b * SLEN + p];
    out[idx] = (bf16_t)((tok == 0) ? 0.0f : embed[(size_t)tok * EDIM + e]);
  }
}

__global__ void cvt_bf16_kernel(const float* __restrict__ src,
                                bf16_t* __restrict__ dst, size_t n) {
  for (size_t i = blockIdx.x * (size_t)blockDim.x + threadIdx.x; i < n;
       i += (size_t)gridDim.x * blockDim.x)
    dst[i] = (bf16_t)src[i];
}

// ---------------------------------------------------------------------------
extern "C" void kernel_launch(void* const* d_in, const int* in_sizes, int n_in,
                              void* d_out, int out_size, void* d_ws, size_t ws_size,
                              hipStream_t stream) {
  (void)in_sizes; (void)n_in; (void)out_size; (void)ws_size;
  const int*   src   = (const int*)d_in[0];
  const int*   tgt   = (const int*)d_in[1];
  const float* embed = (const float*)d_in[2];
  const float* eWih0 = (const float*)d_in[3];
  const float* eWhh0 = (const float*)d_in[4];
  const float* ebih0 = (const float*)d_in[5];
  const float* ebhh0 = (const float*)d_in[6];
  const float* eWih1 = (const float*)d_in[7];
  const float* eWhh1 = (const float*)d_in[8];
  const float* ebih1 = (const float*)d_in[9];
  const float* ebhh1 = (const float*)d_in[10];
  const float* dWih  = (const float*)d_in[11];
  const float* dWhh  = (const float*)d_in[12];
  const float* dbih  = (const float*)d_in[13];
  const float* dbhh  = (const float*)d_in[14];
  const float* fcW   = (const float*)d_in[15];
  const float* fcb   = (const float*)d_in[16];
  float* out = (float*)d_out;

  // ---- workspace carve (all offsets 256B aligned) ----
  uint8_t* ws = (uint8_t*)d_ws;
  size_t off = 0;
  auto alloc = [&](size_t bytes) -> void* {
    void* p = ws + off;
    off = (off + bytes + 255) & ~(size_t)255;
    return p;
  };
  bf16_t* wih0_bf = (bf16_t*)alloc((size_t)GDIM * EDIM * 2);
  bf16_t* whh0_bf = (bf16_t*)alloc((size_t)GDIM * HDIM * 2);
  bf16_t* wih1_bf = (bf16_t*)alloc((size_t)GDIM * HDIM * 2);
  bf16_t* whh1_bf = (bf16_t*)alloc((size_t)GDIM * HDIM * 2);
  bf16_t* dwih_bf = (bf16_t*)alloc((size_t)GDIM * DECIN * 2);
  bf16_t* dwhh_bf = (bf16_t*)alloc((size_t)GDIM * HDIM * 2);
  bf16_t* fcw_bf  = (bf16_t*)alloc((size_t)VOCAB * HDIM * 2);
  bf16_t* x0_bf   = (bf16_t*)alloc((size_t)BATCH * SLEN * EDIM * 2);   // embedded src
  bf16_t* x1_bf   = (bf16_t*)alloc((size_t)BATCH * SLEN * HDIM * 2);   // layer0 outputs
  float*  enc_out = (float*) alloc((size_t)BATCH * SLEN * HDIM * 4);   // layer1 outputs
  bf16_t* demb_bf = (bf16_t*)alloc((size_t)BATCH * DSTEPS * EDIM * 2); // embedded tgt[:, :-1]
  float*  gparts  = (float*) alloc((size_t)KSLICES * BATCH * GDIM * 4);
  float*  c0      = (float*) alloc((size_t)BATCH * HDIM * 4);
  bf16_t* h0_bf   = (bf16_t*)alloc((size_t)BATCH * HDIM * 2);
  float*  c1      = (float*) alloc((size_t)BATCH * HDIM * 4);
  bf16_t* h1_bf   = (bf16_t*)alloc((size_t)BATCH * HDIM * 2);
  float*  h1_f32  = (float*) alloc((size_t)BATCH * HDIM * 4);
  bf16_t* dec_in  = (bf16_t*)alloc((size_t)BATCH * DECIN * 2);

  // ---- one-time (per call) fp32 -> bf16 weight conversion ----
  auto cvt = [&](const float* s, bf16_t* d, size_t n) {
    int blocks = (int)((n + 255) / 256);
    if (blocks > 4096) blocks = 4096;
    cvt_bf16_kernel<<<blocks, 256, 0, stream>>>(s, d, n);
  };
  cvt(eWih0, wih0_bf, (size_t)GDIM * EDIM);
  cvt(eWhh0, whh0_bf, (size_t)GDIM * HDIM);
  cvt(eWih1, wih1_bf, (size_t)GDIM * HDIM);
  cvt(eWhh1, whh1_bf, (size_t)GDIM * HDIM);
  cvt(dWih,  dwih_bf, (size_t)GDIM * DECIN);
  cvt(dWhh,  dwhh_bf, (size_t)GDIM * HDIM);
  cvt(fcW,   fcw_bf,  (size_t)VOCAB * HDIM);

  // ---- embeddings ----
  {
    size_t n = (size_t)BATCH * SLEN * EDIM;
    embed_lookup_kernel<<<2048, 256, 0, stream>>>(src, SLEN, embed, x0_bf, n);
    size_t m = (size_t)BATCH * DSTEPS * EDIM;
    embed_lookup_kernel<<<2048, 256, 0, stream>>>(tgt, DSTEPS, embed, demb_bf, m);
  }

  // ---- zero initial states ----
  hipMemsetAsync(c0,     0, (size_t)BATCH * HDIM * 4, stream);
  hipMemsetAsync(h0_bf,  0, (size_t)BATCH * HDIM * 2, stream);
  hipMemsetAsync(c1,     0, (size_t)BATCH * HDIM * 4, stream);
  hipMemsetAsync(h1_bf,  0, (size_t)BATCH * HDIM * 2, stream);
  hipMemsetAsync(h1_f32, 0, (size_t)BATCH * HDIM * 4, stream);

  const dim3 gatesGrid(64), gatesBlk(256);   // 512 waves (256 N-tiles x 2 K-slices)
  const dim3 pwGrid(128),  pwBlk(256);       // 32768 threads
  const dim3 fcGrid(250),  fcBlk(256);       // 2000 waves (one N-tile each, both M-tiles)

  // ---- encoder layer 0 ----
  for (int t = 0; t < SLEN; ++t) {
    lstm_gates_kernel<<<gatesGrid, gatesBlk, 0, stream>>>(
        x0_bf + (size_t)t * EDIM, (size_t)SLEN * EDIM, EDIM,
        wih0_bf, h0_bf, whh0_bf, gparts);
    lstm_pointwise_kernel<<<pwGrid, pwBlk, 0, stream>>>(
        gparts, ebih0, ebhh0, c0, nullptr, h0_bf,
        x1_bf + (size_t)t * HDIM, nullptr);
  }
  // ---- encoder layer 1 ----
  for (int t = 0; t < SLEN; ++t) {
    lstm_gates_kernel<<<gatesGrid, gatesBlk, 0, stream>>>(
        x1_bf + (size_t)t * HDIM, (size_t)SLEN * HDIM, HDIM,
        wih1_bf, h1_bf, whh1_bf, gparts);
    lstm_pointwise_kernel<<<pwGrid, pwBlk, 0, stream>>>(
        gparts, ebih1, ebhh1, c1, h1_f32, h1_bf,
        nullptr, enc_out + (size_t)t * HDIM);
  }
  // ---- decoder: h1/c1 carry over as initial state ----
  for (int t = 0; t < DSTEPS; ++t) {
    attn_kernel<<<BATCH, 256, 0, stream>>>(enc_out, h1_f32, demb_bf, t, dec_in);
    lstm_gates_kernel<<<gatesGrid, gatesBlk, 0, stream>>>(
        dec_in, DECIN, DECIN, dwih_bf, h1_bf, dwhh_bf, gparts);
    lstm_pointwise_kernel<<<pwGrid, pwBlk, 0, stream>>>(
        gparts, dbih, dbhh, c1, h1_f32, h1_bf, nullptr, nullptr);
    fc_kernel<<<fcGrid, fcBlk, 0, stream>>>(h1_bf, fcw_bf, fcb, out, t);
  }
}
